// Head_43207370998292
// MI455X (gfx1250) — matrix operand order repaired and checked by reading, not compiled, not run on hardware
//
#include <hip/hip_runtime.h>
#include <hip/hip_bf16.h>

typedef __attribute__((ext_vector_type(16))) _Float16 v16h;
typedef __attribute__((ext_vector_type(8)))  _Float16 v8h;
typedef __attribute__((ext_vector_type(8)))  float    v8f;

#define EMB   1024
#define HD    64
#define TSEQ  4096
#define NROWS 16384   // B*T

static __device__ __forceinline__ v8f wmma_f16(v16h a, v16h b, v8f c) {
    return __builtin_amdgcn_wmma_f32_16x16x32_f16(false, a, false, b, (short)0, c,
                                                  false, false);
}
static __device__ __forceinline__ v16h cat8(v8h lo, v8h hi) {
    return __builtin_shufflevector(lo, hi, 0,1,2,3,4,5,6,7,8,9,10,11,12,13,14,15);
}

// ---------------------------------------------------------------------------
// Kernel 0: transpose + f16-convert the three weight matrices.
//   W[1024][64] fp32 -> Wt[m][64][1024] f16   (B-fragments become contiguous)
// ---------------------------------------------------------------------------
__global__ __launch_bounds__(256) void wt_prep(
    const float* __restrict__ Wq, const float* __restrict__ Wk,
    const float* __restrict__ Wv, _Float16* __restrict__ Wt)
{
    const int id  = blockIdx.x * 256 + threadIdx.x;   // 0 .. 3*64K-1
    const int col = id & 63;
    const int kk  = (id >> 6) & 1023;
    const int m   = id >> 16;
    const float* W = (m == 0) ? Wq : (m == 1) ? Wk : Wv;
    Wt[((size_t)m << 16) + ((size_t)col << 10) + kk] = (_Float16)W[(size_t)kk * HD + col];
}

// ---------------------------------------------------------------------------
// Kernel 1: fused QKV projection with double-buffered LDS weight staging.
// The 12 B-fragments per k-step are identical across all 8 waves of a block:
// the block cooperatively stages a 12KB Wt slice (next k-step) into LDS while
// consuming the current one, so the per-wave critical path is a short
// ds_load_b128 instead of an L2 round trip. Global Wt traffic drops 8x.
// ---------------------------------------------------------------------------
static __device__ __forceinline__ void stage_w(_Float16 (*buf)[32],
                                               const _Float16* __restrict__ Wt,
                                               int kk, int tid)
{
#pragma unroll
    for (int j = 0; j < 3; ++j) {
        const int idx  = tid * 3 + j;       // 0..767 : 192 cols x 4 parts
        const int col  = idx >> 2;
        const int part = idx & 3;           // 8 halves (16B) per part
        *(v8h*)&buf[col][part * 8] =
            *(const v8h*)(Wt + ((size_t)col << 10) + kk + part * 8);
    }
}

__global__ __launch_bounds__(256) void qkv_proj(
    const float* __restrict__ x,
    const _Float16* __restrict__ Wt,
    _Float16* __restrict__ q,
    _Float16* __restrict__ k,
    _Float16* __restrict__ vt)   // [B][64][4096]
{
    __shared__ _Float16 wlds[2][192][32];   // 2 x 12 KB slices

    const int tid   = threadIdx.x;
    const int lane  = tid & 31;
    const int wave  = tid >> 5;
    const int l16   = lane & 15;
    const int half  = lane >> 4;
    const int mbase = (blockIdx.x * 8 + wave) * 16;

    v8f acc[12];
#pragma unroll
    for (int i = 0; i < 12; ++i) acc[i] = (v8f){0.f,0.f,0.f,0.f,0.f,0.f,0.f,0.f};

    const float* xrow = x + (size_t)(mbase + l16) * EMB;

    // --- prologue: stage slice for kk = 0, issue first A loads ------------
    stage_w(wlds[0], Wt, 0, tid);
    float4 xa0 = *(const float4*)(xrow + half * 8);
    float4 xa1 = *(const float4*)(xrow + half * 8 + 4);
    float4 xa2 = *(const float4*)(xrow + half * 8 + 16);
    float4 xa3 = *(const float4*)(xrow + half * 8 + 20);
    __syncthreads();

    const int kloc = half ? 8 : 0;
    int pb = 0;
    for (int kk = 0; kk < EMB; kk += 32, pb ^= 1) {
        // --- stage next slice while consuming the current one ------------
        if (kk + 32 < EMB)
            stage_w(wlds[pb ^ 1], Wt, kk + 32, tid);

        // --- convert current A fragment ----------------------------------
        v16h a;
        a[0]=(_Float16)xa0.x; a[1]=(_Float16)xa0.y; a[2]=(_Float16)xa0.z; a[3]=(_Float16)xa0.w;
        a[4]=(_Float16)xa1.x; a[5]=(_Float16)xa1.y; a[6]=(_Float16)xa1.z; a[7]=(_Float16)xa1.w;
        a[8]=(_Float16)xa2.x; a[9]=(_Float16)xa2.y; a[10]=(_Float16)xa2.z; a[11]=(_Float16)xa2.w;
        a[12]=(_Float16)xa3.x; a[13]=(_Float16)xa3.y; a[14]=(_Float16)xa3.z; a[15]=(_Float16)xa3.w;

        // --- issue next A loads (in flight during the WMMA chain) --------
        if (kk + 32 < EMB) {
            const float* xn = xrow + kk + 32 + half * 8;
            xa0 = *(const float4*)(xn);
            xa1 = *(const float4*)(xn + 4);
            xa2 = *(const float4*)(xn + 16);
            xa3 = *(const float4*)(xn + 20);
        }

        // --- 12 WMMAs, B-fragments from LDS (short latency) --------------
#pragma unroll
        for (int i = 0; i < 12; ++i) {
            const v16h bf = cat8(*(const v8h*)&wlds[pb][i * 16 + l16][kloc],
                                 *(const v8h*)&wlds[pb][i * 16 + l16][16 + kloc]);
            acc[i] = wmma_f16(a, bf, acc[i]);
        }
        __syncthreads();
    }

    // --- q, k: row-major stores ------------------------------------------
    _Float16* Os[2] = {q, k};
#pragma unroll
    for (int m = 0; m < 2; ++m) {
        _Float16* o = Os[m];
#pragma unroll
        for (int n = 0; n < 4; ++n) {
            const int col = n * 16 + l16;
#pragma unroll
            for (int r = 0; r < 8; ++r) {
                const int row = mbase + r + half * 8;
                o[(size_t)row * HD + col] = (_Float16)acc[m * 4 + n][r];
            }
        }
    }
    // --- v: transposed store, 8 consecutive t per lane -> one 16B store --
    const int bb = mbase >> 12;                 // batch
    const int t0 = (mbase & 4095) + half * 8;   // t within batch
#pragma unroll
    for (int n = 0; n < 4; ++n) {
        const int col = n * 16 + l16;
        v8h pk;
#pragma unroll
        for (int r = 0; r < 8; ++r) pk[r] = (_Float16)acc[8 + n][r];
        *(v8h*)(vt + ((size_t)bb * HD + col) * TSEQ + t0) = pk;
    }
}

// ---------------------------------------------------------------------------
// Kernel 2: causal flash attention (wave = 16-query tile, key blocks of 32).
// K fragments load as a group before the S WMMAs; V fragments are issued
// right after so their latency hides behind the softmax VALU work.
// ---------------------------------------------------------------------------
__global__ __launch_bounds__(128) void flash_attn(
    const _Float16* __restrict__ q,
    const _Float16* __restrict__ k,
    const _Float16* __restrict__ vt,
    float* __restrict__ out)
{
    __shared__ _Float16 lds[4][16][32];

    const int lane  = threadIdx.x & 31;
    const int wave  = threadIdx.x >> 5;
    const int l16   = lane & 15;
    const int half  = lane >> 4;

    const int    qtile = blockIdx.x * 4 + wave;        // 0..1023
    const int    b     = qtile >> 8;                   // 256 q-tiles per batch
    const int    qloc  = (qtile & 255) * 16;
    const size_t base  = (size_t)b * TSEQ;

    const float scale = 0.03125f;                      // 1024^-0.5

    // --- Q A-fragments: 2 chunks over H=64, vectorized --------------------
    v16h qa[2];
    {
        const _Float16* qp = q + (base + qloc + l16) * HD;
#pragma unroll
        for (int c = 0; c < 2; ++c) {
            const int koff = c * 32 + half * 16;
            qa[c] = cat8(*(const v8h*)(qp + koff), *(const v8h*)(qp + koff + 8));
        }
    }

    v8f o[4];
#pragma unroll
    for (int n = 0; n < 4; ++n) o[n] = (v8f){0.f,0.f,0.f,0.f,0.f,0.f,0.f,0.f};
    float mrow[8], lrow[8];
#pragma unroll
    for (int r = 0; r < 8; ++r) { mrow[r] = -__builtin_inff(); lrow[r] = 0.f; }

    const int kc = half ? 16 : 0;

    for (int kb0 = 0; kb0 <= qloc; kb0 += 32) {
        // prefetch next K block (one cacheline per lane)
        if (kb0 + 32 <= qloc)
            __builtin_prefetch(k + (base + kb0 + 32 + lane) * HD, 0, 3);

        // --- load all 4 K fragments, then 4 S WMMAs ----------------------
        const _Float16* kp0 = k + (base + kb0 + l16)      * HD + kc;
        const _Float16* kp1 = k + (base + kb0 + 16 + l16) * HD + kc;
        const v16h kf00 = *(const v16h*)(kp0);
        const v16h kf01 = *(const v16h*)(kp1);
        const v16h kf10 = *(const v16h*)(kp0 + 32);
        const v16h kf11 = *(const v16h*)(kp1 + 32);

        v8f s0 = (v8f){0.f,0.f,0.f,0.f,0.f,0.f,0.f,0.f};
        v8f s1 = s0;
        s0 = wmma_f16(qa[0], kf00, s0);
        s1 = wmma_f16(qa[0], kf01, s1);
        s0 = wmma_f16(qa[1], kf10, s0);
        s1 = wmma_f16(qa[1], kf11, s1);

        // --- issue V fragment loads now; consumed after softmax ----------
        v16h vb[4];
#pragma unroll
        for (int n = 0; n < 4; ++n)
            vb[n] = *(const v16h*)(vt + ((size_t)b * HD + n * 16 + l16) * TSEQ
                                      + kb0 + kc);

        // --- scale + causal mask + row max -------------------------------
        float p0[8], p1[8], rmax[8];
#pragma unroll
        for (int r = 0; r < 8; ++r) {
            const int qg = qloc + r + half * 8;
            float e0 = s0[r] * scale;
            float e1 = s1[r] * scale;
            if (kb0 + l16      > qg) e0 = -__builtin_inff();
            if (kb0 + 16 + l16 > qg) e1 = -__builtin_inff();
            p0[r] = e0; p1[r] = e1;
            float mx = fmaxf(e0, e1);
            for (int mk = 1; mk < 16; mk <<= 1)
                mx = fmaxf(mx, __shfl_xor(mx, mk, 32));
            rmax[r] = mx;
        }

        // --- online softmax update ---------------------------------------
#pragma unroll
        for (int r = 0; r < 8; ++r) {
            const float mnew = fmaxf(mrow[r], rmax[r]);
            const float corr = __expf(mrow[r] - mnew);
            const float e0   = __expf(p0[r] - mnew);
            const float e1   = __expf(p1[r] - mnew);
            float rs = e0 + e1;
            for (int mk = 1; mk < 16; mk <<= 1)
                rs += __shfl_xor(rs, mk, 32);
            lrow[r] = lrow[r] * corr + rs;
            mrow[r] = mnew;
#pragma unroll
            for (int n = 0; n < 4; ++n) o[n][r] *= corr;
            p0[r] = e0; p1[r] = e1;
        }

        // --- P: C-layout -> LDS -> A-layout (in-wave, DS in order) -------
        _Float16 (*slab)[32] = lds[wave];
#pragma unroll
        for (int r = 0; r < 8; ++r) {
            const int row = r + half * 8;
            slab[row][l16]      = (_Float16)p0[r];
            slab[row][16 + l16] = (_Float16)p1[r];
        }
        asm volatile("s_wait_dscnt 0" ::: "memory");
        const int koff = half * 16;
        const v16h pa = cat8(*(const v8h*)&slab[l16][koff],
                             *(const v8h*)&slab[l16][8 + koff]);
        asm volatile("s_wait_dscnt 0" ::: "memory");

        // --- O += P (16x32) x V (32x64) ----------------------------------
#pragma unroll
        for (int n = 0; n < 4; ++n)
            o[n] = wmma_f16(pa, vb[n], o[n]);
    }

    // --- finalize: O /= l, store fp32 ------------------------------------
#pragma unroll
    for (int r = 0; r < 8; ++r) {
        const float il = 1.f / lrow[r];
        const size_t row = base + qloc + r + half * 8;
#pragma unroll
        for (int n = 0; n < 4; ++n)
            out[row * HD + n * 16 + l16] = o[n][r] * il;
    }
}

// ---------------------------------------------------------------------------
extern "C" void kernel_launch(void* const* d_in, const int* in_sizes, int n_in,
                              void* d_out, int out_size, void* d_ws, size_t ws_size,
                              hipStream_t stream)
{
    (void)in_sizes; (void)n_in; (void)out_size; (void)ws_size;
    const float* x  = (const float*)d_in[0];
    const float* Wq = (const float*)d_in[1];
    const float* Wk = (const float*)d_in[2];
    const float* Wv = (const float*)d_in[3];

    _Float16* q  = (_Float16*)d_ws;               // 2 MB
    _Float16* k  = q  + (size_t)NROWS * HD;       // 2 MB
    _Float16* vt = k  + (size_t)NROWS * HD;       // 2 MB (transposed V)
    _Float16* Wt = vt + (size_t)NROWS * HD;       // 384 KB

    wt_prep  <<<768, 256, 0, stream>>>(Wq, Wk, Wv, Wt);
    qkv_proj <<<128, 256, 0, stream>>>(x, Wt, q, k, vt);
    flash_attn<<<256, 128, 0, stream>>>(q, k, vt, (float*)d_out);
}
